// Attention_8211977470049
// MI455X (gfx1250) — compile-verified, matrix operand order checked by reading, not run on hardware
//
#include <hip/hip_runtime.h>
#include <cstddef>

// ---------------------------------------------------------------------------
// Problem constants (match reference)
// ---------------------------------------------------------------------------
#define Bc    2
#define NQc   1024
#define NKVc  4096
#define DQc   1024
#define Hc    16
#define DHc   64
#define INNERc 1024
#define SCALEc 0.125f   // 64^-0.5

typedef __attribute__((ext_vector_type(16))) __bf16 v16bf;   // 32B = 8 VGPRs
typedef __attribute__((ext_vector_type(8)))  __bf16 bf16x8;  // 16B
typedef __attribute__((ext_vector_type(4)))  __bf16 bf16x4;  // 8B
typedef __attribute__((ext_vector_type(8)))  float  v8f;     // 32B = 8 VGPRs
typedef int int4v __attribute__((__vector_size__(16)));      // 16B int vector

__device__ __forceinline__ __bf16 f2bf(float f) { return (__bf16)f; }

// ---------------------------------------------------------------------------
// Async global->LDS copies (CDNA5 GLOBAL_LOAD_ASYNC_TO_LDS_B128, ASYNCcnt).
// Builtin signature (from hipcc diagnostic): (AS1 int4*, AS3 int4*, Ii, Ii).
// Guarded: falls back to the VGPR-roundtrip staging path if unavailable.
// ---------------------------------------------------------------------------
#if defined(__has_builtin)
#if __has_builtin(__builtin_amdgcn_global_load_async_to_lds_b128) && \
    __has_builtin(__builtin_amdgcn_s_wait_asynccnt)
#define HAVE_ASYNC_LDS 1
#endif
#endif
#ifndef HAVE_ASYNC_LDS
#define HAVE_ASYNC_LDS 0
#endif

#if HAVE_ASYNC_LDS
typedef __attribute__((address_space(1))) int4v* gptr_b128;
typedef __attribute__((address_space(3))) int4v* lptr_b128;

__device__ __forceinline__ void async_copy_b128(const __bf16* g, __bf16* l) {
    __builtin_amdgcn_global_load_async_to_lds_b128(
        (gptr_b128)g, (lptr_b128)l, /*offset=*/0, /*cpol=*/0);
}
__device__ __forceinline__ void async_wait0() {
    __builtin_amdgcn_s_wait_asynccnt(0);
}
#endif

// ---------------------------------------------------------------------------
// DPP16 butterfly reductions across a 16-lane row (pure VALU, no LDS).
//   0xB1  = quad_perm(1,0,3,2)   xor1
//   0x4E  = quad_perm(2,3,0,1)   xor2
//   0x124 = row_ror:4            combine quads
//   0x128 = row_ror:8            combine halves
// ---------------------------------------------------------------------------
template <int CTRL>
__device__ __forceinline__ float dpp_mov(float x) {
    int v = __builtin_bit_cast(int, x);
    int r = __builtin_amdgcn_update_dpp(0, v, CTRL, 0xF, 0xF, true);
    return __builtin_bit_cast(float, r);
}
__device__ __forceinline__ float red_max16(float x) {
    x = fmaxf(x, dpp_mov<0xB1>(x));
    x = fmaxf(x, dpp_mov<0x4E>(x));
    x = fmaxf(x, dpp_mov<0x124>(x));
    x = fmaxf(x, dpp_mov<0x128>(x));
    return x;
}
__device__ __forceinline__ float red_sum16(float x) {
    x += dpp_mov<0xB1>(x);
    x += dpp_mov<0x4E>(x);
    x += dpp_mov<0x124>(x);
    x += dpp_mov<0x128>(x);
    return x;
}

// ---------------------------------------------------------------------------
// WMMA fragment helpers (wave32, ISA 7.12.2 layouts)
// A/B frag (16 rows x 32 K, bf16): lane L: row = L%16, half = L/16
//   element j -> k = kc + (j/8)*16 + half*8 + (j%8)
//   -> two contiguous 16B loads at (row, kc+half*8) and +16
// ---------------------------------------------------------------------------
__device__ __forceinline__ v16bf load_frag(const __bf16* base, int stride,
                                           int kc, int lane) {
    int row  = lane & 15;
    int half = lane >> 4;
    const __bf16* p = base + row * stride + kc + half * 8;
    union { v16bf v; bf16x8 h[2]; } u;
    u.h[0] = *reinterpret_cast<const bf16x8*>(p);
    u.h[1] = *reinterpret_cast<const bf16x8*>(p + 16);
    return u.v;
}

__device__ __forceinline__ v8f wmma_bf16(v16bf a, v16bf b, v8f c) {
    return __builtin_amdgcn_wmma_f32_16x16x32_bf16(
        false, a, false, b, (short)0, c, false, false);
}

// ---------------------------------------------------------------------------
// bf16 WMMA GEMM: C[M,N] = A[M,K](f32) * B[K,N](f32), f32 accum.
// Block tile 128x64, K-step 32, 256 threads = 8 waves (4x2 wave grid).
// Double-buffered LDS, one barrier per K-step; next tile's global loads are
// issued before the compute so VMEM latency hides under WMMA.
// MODE 0: q proj -> bf16 [B,H,NQ,64]*SCALE ; MODE 1: kv -> bf16 k/v
// MODE 2: out proj -> f32 [M,N] + bias[n]
// ---------------------------------------------------------------------------
#define BM 128
#define BN 64
#define BK 32
#define LDA 40   // BK + 8 pad (bf16 elems)
#define LDB 40

template <int MODE>
__global__ __launch_bounds__(256)
void gemm_bf16_wmma(const float* __restrict__ A, const float* __restrict__ Bw,
                    float* __restrict__ outF, const float* __restrict__ bias,
                    __bf16* __restrict__ outA, __bf16* __restrict__ outB,
                    int M, int N, int K) {
    __shared__ __bf16 As[2][BM * LDA];
    __shared__ __bf16 Bs[2][BN * LDB];

    const int tid  = threadIdx.x;
    const int wave = tid >> 5;
    const int lane = tid & 31;
    const int half = lane >> 4;
    const int wm   = wave >> 1;       // 0..3
    const int wn   = wave & 1;        // 0..1
    const int m0   = blockIdx.y * BM;
    const int n0   = blockIdx.x * BN;

    const int arow[4] = { (tid + 0)   >> 3, (tid + 256) >> 3,
                          (tid + 512) >> 3, (tid + 768) >> 3 };
    const int acol    = (tid & 7) << 2;
    const int brow[2] = { (tid + 0) >> 4, (tid + 256) >> 4 };
    const int bcol    = (tid & 15) << 2;

    float4 ra[4], rb[2];

    auto stage_load = [&](int k0) {
        #pragma unroll
        for (int it = 0; it < 4; ++it)
            ra[it] = *reinterpret_cast<const float4*>(
                A + (size_t)(m0 + arow[it]) * K + k0 + acol);
        #pragma unroll
        for (int it = 0; it < 2; ++it)
            rb[it] = *reinterpret_cast<const float4*>(
                Bw + (size_t)(k0 + brow[it]) * N + n0 + bcol);
    };
    auto stage_store = [&](int p) {
        #pragma unroll
        for (int it = 0; it < 4; ++it) {
            bf16x4 pk = { f2bf(ra[it].x), f2bf(ra[it].y),
                          f2bf(ra[it].z), f2bf(ra[it].w) };
            *reinterpret_cast<bf16x4*>(&As[p][arow[it] * LDA + acol]) = pk;
        }
        #pragma unroll
        for (int it = 0; it < 2; ++it) {
            Bs[p][(bcol + 0) * LDB + brow[it]] = f2bf(rb[it].x);
            Bs[p][(bcol + 1) * LDB + brow[it]] = f2bf(rb[it].y);
            Bs[p][(bcol + 2) * LDB + brow[it]] = f2bf(rb[it].z);
            Bs[p][(bcol + 3) * LDB + brow[it]] = f2bf(rb[it].w);
        }
    };

    v8f acc[2][2] = {};
    const int nk = K / BK;

    stage_load(0);
    stage_store(0);

    for (int i = 0; i < nk; ++i) {
        __syncthreads();
        const int p = i & 1;
        if (i + 1 < nk) stage_load((i + 1) * BK);

        v16bf a0 = load_frag(&As[p][(wm * 32 + 0)  * LDA], LDA, 0, lane);
        v16bf a1 = load_frag(&As[p][(wm * 32 + 16) * LDA], LDA, 0, lane);
        v16bf b0 = load_frag(&Bs[p][(wn * 32 + 0)  * LDB], LDB, 0, lane);
        v16bf b1 = load_frag(&Bs[p][(wn * 32 + 16) * LDB], LDB, 0, lane);
        acc[0][0] = wmma_bf16(a0, b0, acc[0][0]);
        acc[0][1] = wmma_bf16(a0, b1, acc[0][1]);
        acc[1][0] = wmma_bf16(a1, b0, acc[1][0]);
        acc[1][1] = wmma_bf16(a1, b1, acc[1][1]);

        if (i + 1 < nk) stage_store(p ^ 1);
    }

    // ---- epilogue: D layout lane holds M = r + 8*half, N = lane%16
    #pragma unroll
    for (int tm = 0; tm < 2; ++tm) {
        #pragma unroll
        for (int tn = 0; tn < 2; ++tn) {
            int mbase = m0 + wm * 32 + tm * 16 + half * 8;
            int ncol  = n0 + wn * 32 + tn * 16 + (lane & 15);
            #pragma unroll
            for (int r = 0; r < 8; ++r) {
                int m = mbase + r;
                float val = acc[tm][tn][r];
                if (MODE == 0) {
                    int b = m / NQc, i2 = m % NQc;
                    int h = ncol / DHc, d = ncol % DHc;
                    outA[((((size_t)b * Hc + h) * NQc + i2) * DHc) + d] =
                        f2bf(val * SCALEc);
                } else if (MODE == 1) {
                    int b = m / NKVc, j = m % NKVc;
                    if (ncol < INNERc) {
                        int h = ncol / DHc, d = ncol % DHc;
                        outA[((((size_t)b * Hc + h) * NKVc + j) * DHc) + d] =
                            f2bf(val);
                    } else {
                        int c = ncol - INNERc;
                        int h = c / DHc, d = c % DHc;
                        outB[((((size_t)b * Hc + h) * NKVc + j) * DHc) + d] =
                            f2bf(val);
                    }
                } else {
                    outF[(size_t)m * N + ncol] = val + bias[ncol];
                }
            }
        }
    }
}

// ---------------------------------------------------------------------------
// Flash attention core.
// grid = (NQ/128, H, B), block = 256 (8 waves), each wave owns 16 Q rows.
// Double-buffered K/V tiles (64 keys), ONE barrier per tile.
//  - K tile: async global->LDS copy (ASYNCcnt) when available; the per-wave
//    s_wait_asynccnt 0 happens BEFORE the barrier so LDS is visible to all.
//  - V tile: VGPR path (needs transposed LDS image [dh][kv]).
//  - P tile: per-wave LDS, no barrier (same-wave LDS RAW ordered by DScnt).
// Softmax reductions are DPP16 butterflies (rows live in 16-lane rows).
// ---------------------------------------------------------------------------
#define LKD 72   // 64 + 8 pad

__global__ __launch_bounds__(256)
void flash_attn_wmma(const __bf16* __restrict__ q,
                     const __bf16* __restrict__ k,
                     const __bf16* __restrict__ v,
                     float* __restrict__ O) {
    __shared__ __bf16 Ks[2][64 * LKD];     // [kv][dh]
    __shared__ __bf16 Vs[2][64 * LKD];     // [dh][kv] (transposed)
    __shared__ __bf16 Ps[8][16 * LKD];     // per-wave 16x64 P tile

    const int tid  = threadIdx.x;
    const int wave = tid >> 5;
    const int lane = tid & 31;
    const int half = lane >> 4;
    const int qblk = blockIdx.x;
    const int h    = blockIdx.y;
    const int bz   = blockIdx.z;

    const __bf16* qg = q + (((size_t)bz * Hc + h) * NQc + qblk * 128 + wave * 16) * DHc;
    const __bf16* kg = k + (((size_t)bz * Hc + h) * NKVc) * DHc;
    const __bf16* vg = v + (((size_t)bz * Hc + h) * NKVc) * DHc;

    // Q fragments are KV-loop invariant
    v16bf aQ0 = load_frag(qg, DHc, 0,  lane);
    v16bf aQ1 = load_frag(qg, DHc, 32, lane);

    // staging coordinates (fixed per thread): two 8-elem chunks each for K,V
    const int srow[2] = { (tid + 0) >> 3, (tid + 256) >> 3 };   // 0..63
    const int sc8     = (tid & 7) << 3;                          // dh base

    bf16x8 rk[2], rv[2];

    // issue loads for tile t (K goes straight to LDS buf when async available)
    auto kv_load = [&](int t, int p) {
        const __bf16* kt = kg + (size_t)t * 64 * DHc;
        const __bf16* vt = vg + (size_t)t * 64 * DHc;
        #pragma unroll
        for (int it = 0; it < 2; ++it) {
#if HAVE_ASYNC_LDS
            async_copy_b128(kt + srow[it] * DHc + sc8,
                            &Ks[p][srow[it] * LKD + sc8]);
#else
            rk[it] = *reinterpret_cast<const bf16x8*>(kt + srow[it] * DHc + sc8);
#endif
            rv[it] = *reinterpret_cast<const bf16x8*>(vt + srow[it] * DHc + sc8);
        }
        if (t + 1 < NKVc / 64)   // L2 prefetch of the tile after next
            __builtin_prefetch(kt + 64 * DHc + tid * 16, 0, 0);
    };
    auto kv_store = [&](int p) {
        #pragma unroll
        for (int it = 0; it < 2; ++it) {
#if !HAVE_ASYNC_LDS
            *reinterpret_cast<bf16x8*>(&Ks[p][srow[it] * LKD + sc8]) = rk[it];
#endif
            #pragma unroll
            for (int i = 0; i < 8; ++i)
                Vs[p][(sc8 + i) * LKD + srow[it]] = rv[it][i];
        }
    };

    v8f o[4] = {};
    float mrow[8], lrow[8];
    #pragma unroll
    for (int r = 0; r < 8; ++r) { mrow[r] = -1e30f; lrow[r] = 0.f; }

    __bf16* Pw = Ps[wave];
    const int NT = NKVc / 64;

    kv_load(0, 0);
    kv_store(0);

    for (int j = 0; j < NT; ++j) {
#if HAVE_ASYNC_LDS
        async_wait0();              // my async K copies into buf p are done
#endif
        __syncthreads();            // ... and everyone's staging is visible
        const int p = j & 1;
        if (j + 1 < NT) kv_load(j + 1, p ^ 1);

        // ---- S = Q K^T : 4 column tiles x K=64 (2 chained WMMA each)
        v8f s[4];
        #pragma unroll
        for (int n = 0; n < 4; ++n) {
            v8f z = {};
            v16bf bk0 = load_frag(&Ks[p][(n * 16) * LKD], LKD, 0,  lane);
            v16bf bk1 = load_frag(&Ks[p][(n * 16) * LKD], LKD, 32, lane);
            z = wmma_bf16(aQ0, bk0, z);
            z = wmma_bf16(aQ1, bk1, z);
            s[n] = z;
        }

        // ---- online softmax (DPP16 butterflies within 16-lane rows)
        float alpha[8], ps[8];
        #pragma unroll
        for (int r = 0; r < 8; ++r) {
            float mx = fmaxf(fmaxf(s[0][r], s[1][r]), fmaxf(s[2][r], s[3][r]));
            mx = red_max16(mx);
            float mn = fmaxf(mrow[r], mx);
            alpha[r] = __expf(mrow[r] - mn);
            mrow[r]  = mn;
            float p0 = __expf(s[0][r] - mn); s[0][r] = p0;
            float p1 = __expf(s[1][r] - mn); s[1][r] = p1;
            float p2 = __expf(s[2][r] - mn); s[2][r] = p2;
            float p3 = __expf(s[3][r] - mn); s[3][r] = p3;
            ps[r] = (p0 + p1) + (p2 + p3);
        }
        #pragma unroll
        for (int r = 0; r < 8; ++r) {
            ps[r] = red_sum16(ps[r]);
            lrow[r] = lrow[r] * alpha[r] + ps[r];
        }
        #pragma unroll
        for (int n = 0; n < 4; ++n)
            #pragma unroll
            for (int r = 0; r < 8; ++r)
                o[n][r] *= alpha[r];

        // ---- P: D-layout -> per-wave LDS (A-layout source); no barrier
        #pragma unroll
        for (int n = 0; n < 4; ++n)
            #pragma unroll
            for (int r = 0; r < 8; ++r)
                Pw[(half * 8 + r) * LKD + n * 16 + (lane & 15)] = f2bf(s[n][r]);

        // ---- O += P V
        v16bf aP0 = load_frag(Pw, LKD, 0,  lane);
        v16bf aP1 = load_frag(Pw, LKD, 32, lane);
        #pragma unroll
        for (int n = 0; n < 4; ++n) {
            v16bf bv0 = load_frag(&Vs[p][(n * 16) * LKD], LKD, 0,  lane);
            v16bf bv1 = load_frag(&Vs[p][(n * 16) * LKD], LKD, 32, lane);
            o[n] = wmma_bf16(aP0, bv0, o[n]);
            o[n] = wmma_bf16(aP1, bv1, o[n]);
        }

        if (j + 1 < NT) kv_store(p ^ 1);
    }

    // ---- write O / l  to [B, NQ, INNER] f32
    float inv[8];
    #pragma unroll
    for (int r = 0; r < 8; ++r)
        inv[r] = 1.0f / lrow[r];
    int qrow_base = qblk * 128 + wave * 16 + half * 8;
    #pragma unroll
    for (int n = 0; n < 4; ++n) {
        int col = h * DHc + n * 16 + (lane & 15);
        #pragma unroll
        for (int r = 0; r < 8; ++r) {
            int row = qrow_base + r;
            O[((size_t)bz * NQc + row) * INNERc + col] = o[n][r] * inv[r];
        }
    }
}

// ---------------------------------------------------------------------------
// Host-side launch
// ---------------------------------------------------------------------------
extern "C" void kernel_launch(void* const* d_in, const int* in_sizes, int n_in,
                              void* d_out, int out_size, void* d_ws, size_t ws_size,
                              hipStream_t stream) {
    (void)in_sizes; (void)n_in; (void)out_size; (void)ws_size;
    const float* x    = (const float*)d_in[0];   // [B,NQ,DQ]
    const float* ctx  = (const float*)d_in[1];   // [B,NKV,DQ]
    const float* Wq   = (const float*)d_in[2];   // [DQ,INNER]
    const float* Wkv  = (const float*)d_in[3];   // [DQ,2*INNER]
    const float* Wout = (const float*)d_in[4];   // [INNER,DQ]
    const float* bout = (const float*)d_in[5];   // [DQ]
    float* out = (float*)d_out;                  // [B,NQ,DQ]

    char* ws = (char*)d_ws;
    __bf16* qbuf = (__bf16*)(ws + 0);                          //  4 MiB
    __bf16* kbuf = (__bf16*)(ws + (size_t)4  * 1024 * 1024);   // 16 MiB
    __bf16* vbuf = (__bf16*)(ws + (size_t)20 * 1024 * 1024);   // 16 MiB
    float*  Obuf = (float*)(ws + (size_t)36 * 1024 * 1024);    //  8 MiB

    gemm_bf16_wmma<0><<<dim3(INNERc / BN, (Bc * NQc) / BM), 256, 0, stream>>>(
        x, Wq, nullptr, nullptr, qbuf, nullptr, Bc * NQc, INNERc, DQc);

    gemm_bf16_wmma<1><<<dim3((2 * INNERc) / BN, (Bc * NKVc) / BM), 256, 0, stream>>>(
        ctx, Wkv, nullptr, nullptr, kbuf, vbuf, Bc * NKVc, 2 * INNERc, DQc);

    flash_attn_wmma<<<dim3(NQc / 128, Hc, Bc), 256, 0, stream>>>(
        qbuf, kbuf, vbuf, Obuf);

    gemm_bf16_wmma<2><<<dim3(DQc / BN, (Bc * NQc) / BM), 256, 0, stream>>>(
        Obuf, Wout, out, bout, nullptr, nullptr, Bc * NQc, DQc, DQc);
}
